// Block_21835613733584
// MI455X (gfx1250) — matrix operand order repaired and checked by reading, not compiled, not run on hardware
//
#include <hip/hip_runtime.h>
#include <hip/hip_bf16.h>

// ---------------------------------------------------------------------------
// Transformer block (B=2, T=2048, C=1024, H=16, D=64) for gfx1250.
// All GEMMs + attention on V_WMMA_F32_16X16X32_BF16 (bf16 in, f32 acc).
//  - GEMM: 128x128 macro tile, double-buffered LDS, 1 barrier / K-step,
//    8 WMMA per wave per K-step.  A-tile staged with async Global->LDS
//    (global_load_async_to_lds_b128, ASYNCcnt), B-tile transposed via regs.
//  - Attention: flash, causal; row-max via DPP, row-sum via WMMA against an
//    all-ones fragment; K tile async->LDS, V tile transposed via regs;
//    fully-masked tiles skipped per wave (wave-uniform branch).
// ---------------------------------------------------------------------------

typedef unsigned int u32;
typedef unsigned short u16;
typedef __attribute__((ext_vector_type(16))) __bf16 v16bf;
typedef __attribute__((ext_vector_type(8)))  float  v8f;

#define B_  2
#define T_  2048
#define C_  1024
#define H_  16
#define Dh_ 64
#define M_  (B_ * T_)

union FragBF { v16bf v; u32 u[8]; };

__device__ __forceinline__ int lane_id() { return threadIdx.x & 31; }

// RNE f32 -> bf16
__device__ __forceinline__ u16 f2bf(float x) {
  union { float f; u32 u; } v; v.f = x;
  u32 r = v.u + 0x7FFFu + ((v.u >> 16) & 1u);
  return (u16)(r >> 16);
}
__device__ __forceinline__ u32 pack2bf(float a, float b) {
  return (u32)f2bf(a) | ((u32)f2bf(b) << 16);
}

__device__ __forceinline__ float gelu_f(float x) {
  float x3 = x * x * x;
  return 0.5f * x * (1.0f + tanhf(0.7978845608f * (x + 0.044715f * x3)));
}

// --- CDNA5 async Global->LDS (tracked with ASYNCcnt) -----------------------
// vdst VGPR holds the LDS byte offset (generic LDS addr truncates to [31:0]).
__device__ __forceinline__ u32 lds_off(const void* p) {
  return (u32)(unsigned long long)p;
}
__device__ __forceinline__ void async_load_b128(u32 lds_byte_off, const void* gaddr) {
  asm volatile("global_load_async_to_lds_b128 %0, %1, off"
               :: "v"(lds_byte_off), "v"(gaddr) : "memory");
}
__device__ __forceinline__ void wait_async0() {
  asm volatile("s_wait_asynccnt 0x0" ::: "memory");
}

// Cross-lane max over each 16-lane group, pure DPP.
#define DPP_FMAX_STEP(v, ctrl)                                                \
  v = fmaxf(v, __builtin_bit_cast(float, __builtin_amdgcn_mov_dpp(           \
                   __builtin_bit_cast(int, v), (ctrl), 0xF, 0xF, true)))
__device__ __forceinline__ float rowmax16(float v) {
  DPP_FMAX_STEP(v, 0xB1);   // quad xor1
  DPP_FMAX_STEP(v, 0x4E);   // quad xor2
  DPP_FMAX_STEP(v, 0x141);  // row_half_mirror
  DPP_FMAX_STEP(v, 0x140);  // row_mirror
  return v;
}

// A fragment (16x32 bf16): lane L -> row L%16; g=L/16; VGPR j<4: K=8g+2j(+1);
// j>=4: K=16+8g+2(j-4)(+1).  `base` is row-major [16][rowStride] halves.
__device__ __forceinline__ v16bf load_frag_a(const u16* base, int rowStride, int koff) {
  int l = lane_id();
  int row = l & 15;
  int g = (l >> 4) & 1;
  const u16* p = base + row * rowStride + koff;
  FragBF f;
#pragma unroll
  for (int j = 0; j < 4; ++j) f.u[j]     = *(const u32*)(p + 8 * g + 2 * j);
#pragma unroll
  for (int j = 0; j < 4; ++j) f.u[4 + j] = *(const u32*)(p + 16 + 8 * g + 2 * j);
  return f.v;
}

// B fragment (32x16, K x N): lane L -> col N=L%16; g=L/16; VGPR j<4: K=8g+2j(+1);
// j>=4: K=16+8g+2(j-4)(+1).  Data staged N-major bt[n][k].
__device__ __forceinline__ v16bf load_frag_b(const u16* bt, int rowStride, int koff) {
  int l = lane_id();
  int n = l & 15;
  int g = (l >> 4) & 1;
  const u16* p = bt + n * rowStride + koff + 8 * g;
  FragBF f;
#pragma unroll
  for (int j = 0; j < 4; ++j) f.u[j]     = *(const u32*)(p + 2 * j);
#pragma unroll
  for (int j = 0; j < 4; ++j) f.u[4 + j] = *(const u32*)(p + 16 + 2 * j);
  return f.v;
}

__device__ __forceinline__ v16bf ones_frag() {
  FragBF f;
#pragma unroll
  for (int j = 0; j < 8; ++j) f.u[j] = 0x3F803F80u;  // bf16 1.0 x2
  return f.v;
}

__device__ __forceinline__ v8f wmma_bf16(v16bf a, v16bf b, v8f c) {
  return __builtin_amdgcn_wmma_f32_16x16x32_bf16(false, a, false, b, (short)0, c, false, false);
}

// ---------------------------------------------------------------------------
// f32 -> bf16 bulk convert (weights)
// ---------------------------------------------------------------------------
__global__ void cvt_bf16(const float* __restrict__ in, u16* __restrict__ out, int n2) {
  int i = blockIdx.x * blockDim.x + threadIdx.x;
  if (i < n2) {
    float2 v = ((const float2*)in)[i];
    ((u32*)out)[i] = pack2bf(v.x, v.y);
  }
}

// ---------------------------------------------------------------------------
// LayerNorm over C=1024, one row per block, bf16 out.
// ---------------------------------------------------------------------------
__global__ __launch_bounds__(256) void ln_bf16(const float* __restrict__ x,
                                               const float* __restrict__ wt,
                                               const float* __restrict__ bs,
                                               u16* __restrict__ out) {
  __shared__ float red[16];
  const int row = blockIdx.x;
  const int tid = threadIdx.x;
  const float4 v = ((const float4*)(x + (size_t)row * C_))[tid];
  float s  = v.x + v.y + v.z + v.w;
  float s2 = v.x * v.x + v.y * v.y + v.z * v.z + v.w * v.w;
#pragma unroll
  for (int m = 1; m < 32; m <<= 1) {
    s  += __shfl_xor(s, m, 32);
    s2 += __shfl_xor(s2, m, 32);
  }
  if ((tid & 31) == 0) { red[tid >> 5] = s; red[8 + (tid >> 5)] = s2; }
  __syncthreads();
  float ts = 0.f, ts2 = 0.f;
#pragma unroll
  for (int i = 0; i < 8; ++i) { ts += red[i]; ts2 += red[8 + i]; }
  const float mu  = ts * (1.0f / C_);
  const float var = ts2 * (1.0f / C_) - mu * mu;
  const float rs  = rsqrtf(var + 1e-5f);
  const float4 wv = ((const float4*)wt)[tid];
  const float4 bv = ((const float4*)bs)[tid];
  u32* op = (u32*)(out + (size_t)row * C_ + tid * 4);
  op[0] = pack2bf((v.x - mu) * rs * wv.x + bv.x, (v.y - mu) * rs * wv.y + bv.y);
  op[1] = pack2bf((v.z - mu) * rs * wv.z + bv.z, (v.w - mu) * rs * wv.w + bv.w);
}

// ---------------------------------------------------------------------------
// WMMA GEMM: out[M,N] = A[M,K](bf16) * W[K,N](bf16) + bias (+epilogue).
// Macro tile 128x128, 256 threads (8 waves): wave = 32 rows x 64 cols.
// ---------------------------------------------------------------------------
enum { EPI_BF16 = 0, EPI_GELU_BF16 = 1, EPI_RESID_F32 = 2 };

template <int EPI>
__global__ __launch_bounds__(256) void gemm_wmma(const u16* __restrict__ A,
                                                 const u16* __restrict__ W,
                                                 const float* __restrict__ bias,
                                                 const float* __restrict__ resid,
                                                 void* __restrict__ out,
                                                 int M, int N, int K) {
  __shared__ u16 sA[2][128 * 32];   // [m][k]
  __shared__ u16 sB[2][128 * 32];   // [n][k] (transposed stage)
  const int tid = threadIdx.x;
  const int w  = tid >> 5;
  const int mb = w & 3;             // 32-row block
  const int nb = w >> 2;            // 64-col block
  const int m0 = blockIdx.y * 128;
  const int n0 = blockIdx.x * 128;

  v8f acc[2][4] = {};
  u32 rb[8];

  // A tile 128x32 (8KB): async Global->LDS, 16B per lane, 2 issues/thread.
  auto async_A = [&](int kk, int p) {
#pragma unroll
    for (int i = 0; i < 2; ++i) {
      int idx = tid + i * 256;                 // 512 chunks of 16B
      int r = idx >> 2;                        // 4 chunks per 64B row
      int c = (idx & 3) * 8;                   // halves offset in row
      async_load_b128(lds_off(sA[p] + (size_t)idx * 8),
                      A + (size_t)(m0 + r) * K + kk + c);
    }
  };
  auto load_B_regs = [&](int kk) {
#pragma unroll
    for (int i = 0; i < 8; ++i) {              // W tile 32x128
      int idx = tid + i * 256, r = idx >> 6, c = idx & 63;
      rb[i] = *(const u32*)(W + (size_t)(kk + r) * N + n0 + c * 2);
    }
  };
  auto store_B = [&](int p) {
#pragma unroll
    for (int i = 0; i < 8; ++i) {
      int idx = tid + i * 256, r = idx >> 6, c = idx & 63;
      sB[p][(c * 2 + 0) * 32 + r] = (u16)(rb[i] & 0xFFFFu);
      sB[p][(c * 2 + 1) * 32 + r] = (u16)(rb[i] >> 16);
    }
  };

  const int nk = K >> 5;
  async_A(0, 0);
  load_B_regs(0);
  store_B(0);
  wait_async0();
  __syncthreads();
  int p = 0;
  for (int it = 0; it < nk; ++it) {
    if (it + 1 < nk) {                         // fill buffer p^1 while computing p
      async_A((it + 1) << 5, p ^ 1);
      load_B_regs((it + 1) << 5);
    }
    const u16* baseA = sA[p] + (mb * 32) * 32;
    v16bf fa0 = load_frag_a(baseA, 32, 0);
    v16bf fa1 = load_frag_a(baseA + 16 * 32, 32, 0);
#pragma unroll
    for (int t = 0; t < 4; ++t) {
      v16bf fb = load_frag_b(sB[p] + (nb * 64 + t * 16) * 32, 32, 0);
      acc[0][t] = wmma_bf16(fa0, fb, acc[0][t]);
      acc[1][t] = wmma_bf16(fa1, fb, acc[1][t]);
    }
    if (it + 1 < nk) store_B(p ^ 1);
    wait_async0();
    __syncthreads();
    p ^= 1;
  }

  const int l = tid & 31;
  const int g = l >> 4;
#pragma unroll
  for (int hb = 0; hb < 2; ++hb) {
#pragma unroll
    for (int t = 0; t < 4; ++t) {
      const int n = n0 + nb * 64 + t * 16 + (l & 15);
      const float bn = bias[n];
#pragma unroll
      for (int j = 0; j < 8; ++j) {
        const int r = m0 + mb * 32 + hb * 16 + 8 * g + j;
        float v = acc[hb][t][j] + bn;
        if (EPI == EPI_GELU_BF16) v = gelu_f(v);
        if (EPI == EPI_RESID_F32) {
          ((float*)out)[(size_t)r * N + n] = v + resid[(size_t)r * N + n];
        } else {
          ((u16*)out)[(size_t)r * N + n] = f2bf(v);
        }
      }
    }
  }
}

// ---------------------------------------------------------------------------
// Flash attention, causal.  Block = one (b,h) x 64 query rows, 4 waves.
// ---------------------------------------------------------------------------
__global__ __launch_bounds__(128) void attn_wmma(const u16* __restrict__ qkv,
                                                 u16* __restrict__ y) {
  __shared__ u16 sQ[64 * 64];          // [q][d]
  __shared__ u16 sK[2][32 * 64];       // [tok][d]
  __shared__ u16 sV[2][64 * 32];       // [d][tok] (transposed)
  __shared__ u16 sP[4][16 * 32];       // per-wave P pad [q][tok]
  const int tid = threadIdx.x;
  const int w  = tid >> 5;
  const int l  = tid & 31;
  const int g  = l >> 4;
  const int ln = l & 15;
  const int bh = blockIdx.y;
  const int b  = bh >> 4;
  const int h  = bh & 15;
  const int q0 = blockIdx.x * 64;
  const u16* qp = qkv + (size_t)b * T_ * (3 * C_) + h * Dh_;
  const u16* kp = qp + C_;
  const u16* vp = qp + 2 * C_;

  for (int i = tid; i < 2048; i += 128) {             // Q tile 64x64
    int r = i >> 5, c = i & 31;
    ((u32*)sQ)[i] = *(const u32*)(qp + (size_t)(q0 + r) * (3 * C_) + c * 2);
  }

  // K tile 32x64 (4KB): async Global->LDS (direct copy), 2 issues/thread.
  auto async_K = [&](int kt, int p) {
#pragma unroll
    for (int i = 0; i < 2; ++i) {
      int idx = tid + i * 128;                 // 256 chunks of 16B
      int r = idx >> 3;                        // 8 chunks per 128B row
      int c = (idx & 7) * 8;                   // halves offset in row
      async_load_b128(lds_off(sK[p] + (size_t)idx * 8),
                      kp + (size_t)(kt + r) * (3 * C_) + c);
    }
  };
  u32 rv[8];
  auto load_V_regs = [&](int kt) {
#pragma unroll
    for (int i = 0; i < 8; ++i) {
      int idx = tid + i * 128, r = idx >> 5, c = idx & 31;
      rv[i] = *(const u32*)(vp + (size_t)(kt + r) * (3 * C_) + c * 2);
    }
  };
  auto store_V = [&](int p) {
#pragma unroll
    for (int i = 0; i < 8; ++i) {
      int idx = tid + i * 128, r = idx >> 5, c = idx & 31;
      sV[p][(c * 2 + 0) * 32 + r] = (u16)(rv[i] & 0xFFFFu);
      sV[p][(c * 2 + 1) * 32 + r] = (u16)(rv[i] >> 16);
    }
  };

  async_K(0, 0);
  load_V_regs(0);
  store_V(0);
  wait_async0();
  __syncthreads();
  const v16bf aQ0 = load_frag_a(sQ + w * 16 * 64, 64, 0);
  const v16bf aQ1 = load_frag_a(sQ + w * 16 * 64, 64, 32);
  const v16bf bOne = ones_frag();

  float m_[8];
#pragma unroll
  for (int j = 0; j < 8; ++j) m_[j] = -1e30f;
  v8f acc[4] = {};
  v8f accL = {};

  const int ntiles = (q0 + 64) >> 5;
  const int wave_last = q0 + w * 16 + 15;      // last unmasked key for this wave
  int p = 0;
  for (int it = 0; it < ntiles; ++it) {
    const int kt = it << 5;
    if (it + 1 < ntiles) {
      async_K(kt + 32, p ^ 1);
      load_V_regs(kt + 32);
    }

    if (kt <= wave_last) {                     // wave-uniform: EXEC stays full
      // scores for both 16-key halves of this 32-key tile
      v8f S0 = {}, S1 = {};
      S0 = wmma_bf16(aQ0, load_frag_b(sK[p], 64, 0),  S0);
      S0 = wmma_bf16(aQ1, load_frag_b(sK[p], 64, 32), S0);
      S1 = wmma_bf16(aQ0, load_frag_b(sK[p] + 16 * 64, 64, 0),  S1);
      S1 = wmma_bf16(aQ1, load_frag_b(sK[p] + 16 * 64, 64, 32), S1);

      const int kg0 = kt + ln;
      const int kg1 = kt + 16 + ln;
#pragma unroll
      for (int j = 0; j < 8; ++j) {
        const int qg = q0 + w * 16 + 8 * g + j;
        float v0 = (kg0 <= qg) ? S0[j] * 0.125f : -1e30f;   // 1/sqrt(64) + mask
        float v1 = (kg1 <= qg) ? S1[j] * 0.125f : -1e30f;
        const float mx = rowmax16(fmaxf(v0, v1));
        const float mn = fmaxf(m_[j], mx);
        const float al = __expf(m_[j] - mn);
        m_[j] = mn;
        const float p0 = __expf(v0 - mn);
        const float p1 = __expf(v1 - mn);
        accL[j] *= al;
#pragma unroll
        for (int t = 0; t < 4; ++t) acc[t][j] *= al;
        sP[w][(8 * g + j) * 32 + ln]      = f2bf(p0);
        sP[w][(8 * g + j) * 32 + 16 + ln] = f2bf(p1);
      }
      const v16bf aP = load_frag_a(sP[w], 32, 0);
      accL = wmma_bf16(aP, bOne, accL);                    // row sums via WMMA
#pragma unroll
      for (int t = 0; t < 4; ++t) {
        v16bf bv = load_frag_b(sV[p] + (t * 16) * 32, 32, 0);
        acc[t] = wmma_bf16(aP, bv, acc[t]);
      }
    }

    if (it + 1 < ntiles) store_V(p ^ 1);
    wait_async0();
    __syncthreads();
    p ^= 1;
  }

#pragma unroll
  for (int t = 0; t < 4; ++t) {
#pragma unroll
    for (int j = 0; j < 8; ++j) {
      const int r = q0 + w * 16 + 8 * g + j;
      const int d = t * 16 + ln;
      y[((size_t)b * T_ + r) * C_ + h * Dh_ + d] = f2bf(acc[t][j] / accL[j]);
    }
  }
}

// ---------------------------------------------------------------------------
// Launch
// ---------------------------------------------------------------------------
extern "C" void kernel_launch(void* const* d_in, const int* in_sizes, int n_in,
                              void* d_out, int out_size, void* d_ws, size_t ws_size,
                              hipStream_t stream) {
  (void)in_sizes; (void)n_in; (void)out_size; (void)ws_size;
  const float* x      = (const float*)d_in[0];
  const float* ln1_w  = (const float*)d_in[1];
  const float* ln1_b  = (const float*)d_in[2];
  const float* w_attn = (const float*)d_in[3];
  const float* b_attn = (const float*)d_in[4];
  const float* w_proj = (const float*)d_in[5];
  const float* b_proj = (const float*)d_in[6];
  const float* ln2_w  = (const float*)d_in[7];
  const float* ln2_b  = (const float*)d_in[8];
  const float* w_fc   = (const float*)d_in[9];
  const float* b_fc   = (const float*)d_in[10];
  const float* w_mlp  = (const float*)d_in[11];
  const float* b_mlp  = (const float*)d_in[12];

  char* ws = (char*)d_ws;
  size_t off = 0;
  auto alloc = [&](size_t bytes) {
    void* p = ws + off;
    off += (bytes + 255) & ~(size_t)255;
    return p;
  };
  u16*   wattn_bf = (u16*)alloc((size_t)C_ * 3 * C_ * 2);
  u16*   wproj_bf = (u16*)alloc((size_t)C_ * C_ * 2);
  u16*   wfc_bf   = (u16*)alloc((size_t)C_ * 4 * C_ * 2);
  u16*   wmlp_bf  = (u16*)alloc((size_t)4 * C_ * C_ * 2);
  u16*   h_bf     = (u16*)alloc((size_t)M_ * C_ * 2);
  u16*   qkv_bf   = (u16*)alloc((size_t)M_ * 3 * C_ * 2);
  u16*   y_bf     = (u16*)alloc((size_t)M_ * C_ * 2);
  float* x1       = (float*)alloc((size_t)M_ * C_ * 4);
  u16*   h2_bf    = (u16*)alloc((size_t)M_ * C_ * 2);
  u16*   g_bf     = (u16*)alloc((size_t)M_ * 4 * C_ * 2);

  auto cvt = [&](const float* src, u16* dst, size_t n) {
    int n2 = (int)(n / 2);
    cvt_bf16<<<(n2 + 255) / 256, 256, 0, stream>>>(src, dst, n2);
  };
  cvt(w_attn, wattn_bf, (size_t)C_ * 3 * C_);
  cvt(w_proj, wproj_bf, (size_t)C_ * C_);
  cvt(w_fc,   wfc_bf,   (size_t)C_ * 4 * C_);
  cvt(w_mlp,  wmlp_bf,  (size_t)4 * C_ * C_);

  // h = LN1(x)
  ln_bf16<<<M_, 256, 0, stream>>>(x, ln1_w, ln1_b, h_bf);
  // qkv = h @ w_attn + b_attn
  gemm_wmma<EPI_BF16><<<dim3(3 * C_ / 128, M_ / 128), 256, 0, stream>>>(
      h_bf, wattn_bf, b_attn, nullptr, qkv_bf, M_, 3 * C_, C_);
  // y = causal attention(qkv)
  attn_wmma<<<dim3(T_ / 64, B_ * H_), 128, 0, stream>>>(qkv_bf, y_bf);
  // x1 = x + y @ w_proj + b_proj
  gemm_wmma<EPI_RESID_F32><<<dim3(C_ / 128, M_ / 128), 256, 0, stream>>>(
      y_bf, wproj_bf, b_proj, x, x1, M_, C_, C_);
  // h2 = LN2(x1)
  ln_bf16<<<M_, 256, 0, stream>>>(x1, ln2_w, ln2_b, h2_bf);
  // g = gelu(h2 @ w_fc + b_fc)
  gemm_wmma<EPI_GELU_BF16><<<dim3(4 * C_ / 128, M_ / 128), 256, 0, stream>>>(
      h2_bf, wfc_bf, b_fc, nullptr, g_bf, M_, 4 * C_, C_);
  // out = x1 + g @ w_mlp_proj + b_mlp_proj
  gemm_wmma<EPI_RESID_F32><<<dim3(C_ / 128, M_ / 128), 256, 0, stream>>>(
      g_bf, wmlp_bf, b_mlp, x1, (float*)d_out, M_, C_, 4 * C_);
}
